// AttnDecoder_14740327759918
// MI455X (gfx1250) — compile-verified, matrix operand order checked by reading, not run on hardware
//
#include <hip/hip_runtime.h>
#include <hip/hip_bf16.h>

// ---------------- CDNA5 / gfx1250 WMMA types ----------------
typedef __bf16 bf16;
typedef __attribute__((ext_vector_type(16))) __bf16 v16bf;
typedef __attribute__((ext_vector_type(8)))  float  v8f;
typedef __attribute__((ext_vector_type(4)))  float  f32x4;

#define NCLS 32000
#define EMB  512
#define HID  512
#define BSZ  32
#define TENC 128
#define TDEC 64
#define G4H  2048   // 4*H
#define H2   1024   // 2*H
#define DIN  3072   // 2*H*L + 2*H
#define KIN0 1536   // E + 2H (layer0 input width)
#define KCAT 1536   // layer1: [x(1024) | h(512)]
#define MROWS 2048  // TDEC*BSZ

static __device__ __forceinline__ float sigf(float x) { return 1.0f / (1.0f + __expf(-x)); }

// Load one 16x32 bf16 operand fragment (A or B) for v_wmma_f32_16x16x32_bf16.
// Per ISA layout: lanes 0-15 hold K=[0..7] and K=[16..23]; lanes 16-31 hold
// K=[8..15] and K=[24..31] of row (m = lane&15). Two 16B loads per lane.
static __device__ __forceinline__ v16bf load_frag(const bf16* __restrict__ p, int k0, int hi) {
  union { f32x4 f[2]; v16bf v; } u;
  u.f[0] = *(const f32x4*)(p + k0 + hi * 8);
  u.f[1] = *(const f32x4*)(p + k0 + 16 + hi * 8);
  return u.v;
}

static __device__ __forceinline__ v8f wmma_bf(v16bf a, v16bf b, v8f c) {
  // (neg_a, A, neg_b, B, c_mod, C, reuse_a, reuse_b)
  return __builtin_amdgcn_wmma_f32_16x16x32_bf16(false, a, false, b, (short)0, c, false, false);
}

// ---- CDNA5 async global->LDS copy (ASYNCcnt-tracked, §08_async_tensor) ----
// Each lane moves 16 bytes: LDS[lds_off] = MEM[gaddr]. One instruction per wave.
// lds_off must be derived from the real shared-object address (ptrtoint) so the
// object escapes and the compiler keeps the consumer-side ds_loads live.
static __device__ __forceinline__ void async_b128(unsigned lds_off, const bf16* gaddr) {
  asm volatile("global_load_async_to_lds_b128 %0, %1, off"
               :: "v"(lds_off), "v"(gaddr) : "memory");
}
static __device__ __forceinline__ void wait_async_le1() {
  asm volatile("s_wait_asynccnt 0x1" ::: "memory");
}
static __device__ __forceinline__ void wait_async_le0() {
  asm volatile("s_wait_asynccnt 0x0" ::: "memory");
}

// ---------------- attention collapse (exact: softmax shift-invariance) ----------------
// v_enc[j] = sum_k attn2_W[k] * attn1_W[k, j], j < 2H (encoder-feature part only)
__global__ void venc_kernel(const float* __restrict__ a1W, const float* __restrict__ a2W,
                            float* __restrict__ venc) {
  int j = blockIdx.x * blockDim.x + threadIdx.x;
  if (j >= H2) return;
  float s = 0.f;
  for (int k = 0; k < G4H; ++k) s += a2W[k] * a1W[(size_t)k * DIN + j];
  venc[j] = s;
}

// sc[b,t] = enc_output[b,t,:] . v_enc   (h-part and biases are constant over t)
__global__ void score_kernel(const float* __restrict__ enc, const float* __restrict__ venc,
                             float* __restrict__ sc) {
  __shared__ float sm[256];
  int bt = blockIdx.x;
  const float* row = enc + (size_t)bt * H2;
  float s = 0.f;
  for (int k = threadIdx.x; k < H2; k += 256) s += row[k] * venc[k];
  sm[threadIdx.x] = s;
  __syncthreads();
  for (int w = 128; w > 0; w >>= 1) {
    if (threadIdx.x < w) sm[threadIdx.x] += sm[threadIdx.x + w];
    __syncthreads();
  }
  if (threadIdx.x == 0) sc[bt] = sm[0];
}

// softmax over t, ctx[b,:] = sum_t w[b,t] * enc[b,t,:]
__global__ void ctx_kernel(const float* __restrict__ enc, const float* __restrict__ sc,
                           float* __restrict__ ctx) {
  __shared__ float sw[TENC];
  int b = blockIdx.x;
  float mx = -1e30f;
  for (int t = 0; t < TENC; ++t) mx = fmaxf(mx, sc[b * TENC + t]);
  float den = 0.f;
  for (int t = 0; t < TENC; ++t) den += __expf(sc[b * TENC + t] - mx);
  if (threadIdx.x < TENC) sw[threadIdx.x] = __expf(sc[b * TENC + threadIdx.x] - mx) / den;
  __syncthreads();
  for (int j = threadIdx.x; j < H2; j += blockDim.x) {
    float a = 0.f;
    for (int t = 0; t < TENC; ++t) a += sw[t] * enc[((size_t)b * TENC + t) * H2 + j];
    ctx[b * H2 + j] = a;
  }
}

// pre0[d][b][g] = ctx[b] . Wih0[d][g][512:1536] + bih0[d][g] + bhh0[d][g]
__global__ void pre0_kernel(const float* __restrict__ ctx, const float* __restrict__ Wih0,
                            const float* __restrict__ bih0, const float* __restrict__ bhh0,
                            float* __restrict__ pre0) {
  int idx = blockIdx.x * blockDim.x + threadIdx.x;
  if (idx >= 2 * BSZ * G4H) return;
  int d = idx / (BSZ * G4H);
  int rem = idx % (BSZ * G4H);
  int b = rem / G4H, g = rem % G4H;
  const float* w = Wih0 + ((size_t)d * G4H + g) * KIN0 + EMB;
  const float* c = ctx + (size_t)b * H2;
  float s = bih0[d * G4H + g] + bhh0[d * G4H + g];
  for (int k = 0; k < H2; ++k) s += c[k] * w[k];
  pre0[idx] = s;
}

__global__ void bias1_kernel(const float* __restrict__ bih1, const float* __restrict__ bhh1,
                             float* __restrict__ bias1) {
  int i = blockIdx.x * blockDim.x + threadIdx.x;
  if (i < 2 * G4H) bias1[i] = bih1[i] + bhh1[i];
}

// ---------------- bf16 packing ----------------
__global__ void cvt_kernel(const float* __restrict__ src, bf16* __restrict__ dst, long n) {
  long i = (long)blockIdx.x * blockDim.x + threadIdx.x;
  long st = (long)gridDim.x * blockDim.x;
  for (; i < n; i += st) dst[i] = (bf16)src[i];
}

// Wcat1[d] = [Wih1[d] (2048x1024) | Whh1[d] (2048x512)] -> (2048 x 1536) bf16
__global__ void build_wcat1_kernel(const float* __restrict__ Wih1, const float* __restrict__ Whh1,
                                   bf16* __restrict__ dst) {
  long n = 2L * G4H * KCAT;
  long i = (long)blockIdx.x * blockDim.x + threadIdx.x;
  long st = (long)gridDim.x * blockDim.x;
  for (; i < n; i += st) {
    int d = (int)(i / ((long)G4H * KCAT));
    long rem = i % ((long)G4H * KCAT);
    int g = (int)(rem / KCAT), k = (int)(rem % KCAT);
    float v = (k < H2) ? Wih1[((size_t)d * G4H + g) * H2 + k]
                       : Whh1[((size_t)d * G4H + g) * HID + (k - H2)];
    dst[i] = (bf16)v;
  }
}

// We0[d] = Wih0[d][:, 0:512] -> (2048 x 512) bf16 (embedding part)
__global__ void build_we0_kernel(const float* __restrict__ Wih0, bf16* __restrict__ dst) {
  long n = 2L * G4H * EMB;
  long i = (long)blockIdx.x * blockDim.x + threadIdx.x;
  long st = (long)gridDim.x * blockDim.x;
  for (; i < n; i += st) {
    int d = (int)(i / ((long)G4H * EMB));
    long rem = i % ((long)G4H * EMB);
    int g = (int)(rem / EMB), k = (int)(rem % EMB);
    dst[i] = (bf16)Wih0[((size_t)d * G4H + g) * KIN0 + k];
  }
}

// Xe[(t*B+b), 0:512] = emb[input[b,t]]
__global__ void gather_emb_kernel(const int* __restrict__ input, const float* __restrict__ emb,
                                  bf16* __restrict__ Xe) {
  long n = (long)MROWS * EMB;
  long i = (long)blockIdx.x * blockDim.x + threadIdx.x;
  long st = (long)gridDim.x * blockDim.x;
  for (; i < n; i += st) {
    int row = (int)(i / EMB), e = (int)(i % EMB);
    int t = row >> 5, b = row & 31;
    Xe[i] = (bf16)emb[(size_t)input[b * TDEC + t] * EMB + e];
  }
}

// x1[b, d*512+j] = enc_h0[d,b,j]; h1 from layers 2+d; c0/c1 f32 copies
__global__ void init_state_kernel(const float* __restrict__ h0, const float* __restrict__ c0i,
                                  bf16* __restrict__ x1, bf16* __restrict__ h1,
                                  float* __restrict__ c0, float* __restrict__ c1) {
  int i = blockIdx.x * blockDim.x + threadIdx.x;
  if (i >= BSZ * H2) return;
  int b = i / H2, col = i % H2;
  int d = col >> 9, j = col & 511;
  x1[i] = (bf16)h0[((size_t)d * BSZ + b) * HID + j];
  h1[i] = (bf16)h0[((size_t)(2 + d) * BSZ + b) * HID + j];
  c0[i] = c0i[((size_t)d * BSZ + b) * HID + j];
  c1[i] = c0i[((size_t)(2 + d) * BSZ + b) * HID + j];
}

// ---------------- WMMA GEMM: xe_g[d] = Xe (2048x512) @ We0[d]^T -> (2048x2048) f32 ----------------
__global__ void xe_gemm_kernel(const bf16* __restrict__ A, const bf16* __restrict__ Wb,
                               float* __restrict__ C) {
  const int K = EMB;
  int d = blockIdx.z;
  int lane = threadIdx.x & 31, wv = threadIdx.x >> 5;
  int hi = lane >> 4, ln = lane & 15;
  int mt = blockIdx.y * 128 + wv * 16;
  int n0 = blockIdx.x * 64;
  const bf16* arow = A + (size_t)(mt + ln) * K;
  v8f acc[4] = {};
  for (int k0 = 0; k0 < K; k0 += 32) {
    v16bf af = load_frag(arow, k0, hi);
#pragma unroll
    for (int nt = 0; nt < 4; ++nt) {
      const bf16* wrow = Wb + ((size_t)d * G4H + n0 + nt * 16 + ln) * K;
      acc[nt] = wmma_bf(af, load_frag(wrow, k0, hi), acc[nt]);
    }
  }
#pragma unroll
  for (int nt = 0; nt < 4; ++nt)
#pragma unroll
    for (int r = 0; r < 8; ++r) {
      int m = mt + r + hi * 8;
      int n = n0 + nt * 16 + ln;
      C[(size_t)d * MROWS * G4H + (size_t)m * G4H + n] = acc[nt][r];
    }
}

// ---------------- recurrent step: layer0 cells (both directions), fused GEMM+LSTM ----------------
// 32 blocks x 128 threads = 128 waves: d(2) x mtile(2) x jgroup(32).
// Each wave computes gate columns {j0, j0+512, j0+1024, j0+1536} so i/f/g/o fuse in registers.
__global__ void step_l0_kernel(const bf16* __restrict__ x1c, bf16* __restrict__ x1n,
                               float* __restrict__ c0,
                               const bf16* __restrict__ Whh0b,   // (2,2048,512) bf16
                               const float* __restrict__ xeg,    // (2,2048,2048) f32
                               const float* __restrict__ pre0,   // (2,32,2048) f32
                               int t) {
  int lane = threadIdx.x & 31, wv = threadIdx.x >> 5;
  int wid = blockIdx.x * 4 + wv;
  int d = wid >> 6;
  int mt = ((wid >> 5) & 1) << 4;
  int j0 = (wid & 31) << 4;
  int hi = lane >> 4, ln = lane & 15;
  const bf16* arow = x1c + (size_t)(mt + ln) * H2 + d * HID;
  v8f acc[4] = {};
  for (int k0 = 0; k0 < HID; k0 += 32) {
    v16bf af = load_frag(arow, k0, hi);
#pragma unroll
    for (int gc = 0; gc < 4; ++gc) {
      const bf16* wrow = Whh0b + (size_t)(d * G4H + gc * HID + j0 + ln) * HID;
      acc[gc] = wmma_bf(af, load_frag(wrow, k0, hi), acc[gc]);
    }
  }
  int j = j0 + ln;
#pragma unroll
  for (int r = 0; r < 8; ++r) {
    int b = mt + r + hi * 8;
    const float* xr = xeg + (size_t)d * MROWS * G4H + (size_t)(t * BSZ + b) * G4H;
    const float* pr = pre0 + (size_t)d * BSZ * G4H + (size_t)b * G4H;
    float gi = acc[0][r] + xr[j] + pr[j];
    float gf = acc[1][r] + xr[j + 512] + pr[j + 512];
    float gg = acc[2][r] + xr[j + 1024] + pr[j + 1024];
    float go = acc[3][r] + xr[j + 1536] + pr[j + 1536];
    int ci = b * H2 + d * HID + j;
    float cn = sigf(gf) * c0[ci] + sigf(gi) * tanhf(gg);
    c0[ci] = cn;
    x1n[ci] = (bf16)(sigf(go) * tanhf(cn));
  }
}

// layer1 cells: u = [x1_new(1024) | h1_prev_dir(512)], W = Wcat1[d] (2048x1536)
__global__ void step_l1_kernel(const bf16* __restrict__ x1new, const bf16* __restrict__ h1c,
                               bf16* __restrict__ h1n, float* __restrict__ c1,
                               const bf16* __restrict__ Wcat1b,  // (2,2048,1536) bf16
                               const float* __restrict__ bias1,  // (2,2048) f32
                               bf16* __restrict__ decb,          // (2048,1024) bf16
                               int t) {
  int lane = threadIdx.x & 31, wv = threadIdx.x >> 5;
  int wid = blockIdx.x * 4 + wv;
  int d = wid >> 6;
  int mt = ((wid >> 5) & 1) << 4;
  int j0 = (wid & 31) << 4;
  int hi = lane >> 4, ln = lane & 15;
  const bf16* ax = x1new + (size_t)(mt + ln) * H2;
  const bf16* ah = h1c + (size_t)(mt + ln) * H2 + d * HID;
  v8f acc[4] = {};
  for (int k0 = 0; k0 < KCAT; k0 += 32) {
    const bf16* ap = (k0 < H2) ? (ax + k0) : (ah + (k0 - H2));
    v16bf af = load_frag(ap, 0, hi);
#pragma unroll
    for (int gc = 0; gc < 4; ++gc) {
      const bf16* wrow = Wcat1b + (size_t)(d * G4H + gc * HID + j0 + ln) * KCAT;
      acc[gc] = wmma_bf(af, load_frag(wrow, k0, hi), acc[gc]);
    }
  }
  int j = j0 + ln;
#pragma unroll
  for (int r = 0; r < 8; ++r) {
    int b = mt + r + hi * 8;
    const float* bs = bias1 + (size_t)d * G4H;
    float gi = acc[0][r] + bs[j];
    float gf = acc[1][r] + bs[j + 512];
    float gg = acc[2][r] + bs[j + 1024];
    float go = acc[3][r] + bs[j + 1536];
    int ci = b * H2 + d * HID + j;
    float cn = sigf(gf) * c1[ci] + sigf(gi) * tanhf(gg);
    c1[ci] = cn;
    float h = sigf(go) * tanhf(cn);
    h1n[ci] = (bf16)h;
    decb[(size_t)(t * BSZ + b) * H2 + d * HID + j] = (bf16)h;
  }
}

// ---------------- final projection: (2048 x 32000 x 1024) WMMA GEMM + bias ----------------
// grid (500, 16), block 256 (8 waves). All 8 waves share columns n0..n0+63, so the
// 64x32 bf16 weight tile (4KB) is staged in LDS once per block via CDNA5 async
// global->LDS copies (double-buffered, ASYNCcnt-gated). LDS offsets for the async
// writes are taken from the shared object itself (ptrtoint of a flat LDS pointer:
// low 32 bits are the wave-relative LDS address per ISA §10.2), which also escapes
// the object so the consumer-side ds_load_b128 reads stay live.
__global__ void out_gemm_kernel(const bf16* __restrict__ dec, const bf16* __restrict__ Wo,
                                const float* __restrict__ ob, float* __restrict__ out) {
  // [buf][row][chunk]: row = local weight row (64), chunk = 4 x 16B = 32 bf16 of K
  __shared__ f32x4 smemB4[2][64][4];
  int tid = threadIdx.x;
  int lane = tid & 31, wv = tid >> 5;
  int hi = lane >> 4, ln = lane & 15;
  int mt = blockIdx.y * 128 + wv * 16;
  int n0 = blockIdx.x * 64;

  // async-fill mapping: thread tid moves 16B of weight row (n0 + tid/4)
  int frow = tid >> 2;               // 0..63 local weight row
  int fchunk = tid & 3;              // 16B chunk within the 32-wide k-slice
  const bf16* wbase = Wo + (size_t)(n0 + frow) * H2 + fchunk * 8;
  unsigned lds0 = (unsigned)(uintptr_t)&smemB4[0][frow][fchunk];   // escapes smemB4
  unsigned bufbytes = (unsigned)((uintptr_t)&smemB4[1][0][0] -
                                 (uintptr_t)&smemB4[0][0][0]);     // 4KB

  const bf16* arow = dec + (size_t)(mt + ln) * H2;
  v8f acc[4] = {};

  async_b128(lds0, wbase);  // prefetch k-tile 0 into buf0
  for (int kt = 0; kt < 32; ++kt) {
    int buf = kt & 1;
    if (kt + 1 < 32) {
      async_b128(lds0 + (buf ^ 1) * bufbytes, wbase + (kt + 1) * 32);
      wait_async_le1();               // oldest (current buf) fill complete
    } else {
      wait_async_le0();
    }
    __syncthreads();                  // tile visible to all 8 waves
    v16bf af = load_frag(arow, kt * 32, hi);
#pragma unroll
    for (int nt = 0; nt < 4; ++nt) {
      int lrow = nt * 16 + ln;
      union { f32x4 f[2]; v16bf v; } u;
      u.f[0] = smemB4[buf][lrow][hi];      // K chunk [hi*8 .. hi*8+7]
      u.f[1] = smemB4[buf][lrow][2 + hi];  // K chunk [16+hi*8 .. 16+hi*8+7]
      acc[nt] = wmma_bf(af, u.v, acc[nt]);
    }
    __syncthreads();                  // all waves done reading before refill
  }

#pragma unroll
  for (int nt = 0; nt < 4; ++nt)
#pragma unroll
    for (int r = 0; r < 8; ++r) {
      int m = mt + r + hi * 8;
      int tt = m >> 5, b = m & 31;
      int n = n0 + nt * 16 + ln;
      out[(size_t)b * TDEC * NCLS + (size_t)tt * NCLS + n] = acc[nt][r] + ob[n];
    }
}

// ---------------- host-side orchestration ----------------
extern "C" void kernel_launch(void* const* d_in, const int* in_sizes, int n_in,
                              void* d_out, int out_size, void* d_ws, size_t ws_size,
                              hipStream_t stream) {
  (void)in_sizes; (void)n_in; (void)out_size; (void)ws_size;
  const int*   input  = (const int*)  d_in[0];
  const float* encout = (const float*)d_in[1];
  const float* enc_h0 = (const float*)d_in[2];
  const float* enc_c0 = (const float*)d_in[3];
  const float* emb    = (const float*)d_in[4];
  const float* a1W    = (const float*)d_in[5];
  // d_in[6] attn1_b, d_in[8] attn2_b: provably cancel in softmax (shift invariance)
  const float* a2W    = (const float*)d_in[7];
  const float* Wih0   = (const float*)d_in[9];
  const float* Whh0   = (const float*)d_in[10];
  const float* bih0   = (const float*)d_in[11];
  const float* bhh0   = (const float*)d_in[12];
  const float* Wih1   = (const float*)d_in[13];
  const float* Whh1   = (const float*)d_in[14];
  const float* bih1   = (const float*)d_in[15];
  const float* bhh1   = (const float*)d_in[16];
  const float* outW   = (const float*)d_in[17];
  const float* outb   = (const float*)d_in[18];
  float* out = (float*)d_out;

  char* ws = (char*)d_ws;
  size_t off = 0;
  auto alloc = [&](size_t bytes) -> void* {
    off = (off + 255) & ~(size_t)255;
    void* p = ws + off;
    off += bytes;
    return p;
  };
  float* venc   = (float*)alloc((size_t)H2 * 4);
  float* scores = (float*)alloc((size_t)BSZ * TENC * 4);
  float* ctx    = (float*)alloc((size_t)BSZ * H2 * 4);
  float* pre0   = (float*)alloc(2ull * BSZ * G4H * 4);
  float* bias1  = (float*)alloc(2ull * G4H * 4);
  bf16*  Whh0b  = (bf16*) alloc(2ull * G4H * HID * 2);
  bf16*  Wcat1b = (bf16*) alloc(2ull * G4H * KCAT * 2);
  bf16*  We0b   = (bf16*) alloc(2ull * G4H * EMB * 2);
  bf16*  Xe     = (bf16*) alloc((size_t)MROWS * EMB * 2);
  float* xeg    = (float*)alloc(2ull * MROWS * G4H * 4);
  bf16*  x1a    = (bf16*) alloc((size_t)BSZ * H2 * 2);
  bf16*  x1b    = (bf16*) alloc((size_t)BSZ * H2 * 2);
  bf16*  h1a    = (bf16*) alloc((size_t)BSZ * H2 * 2);
  bf16*  h1b    = (bf16*) alloc((size_t)BSZ * H2 * 2);
  float* c0     = (float*)alloc((size_t)BSZ * H2 * 4);
  float* c1     = (float*)alloc((size_t)BSZ * H2 * 4);
  bf16*  decb   = (bf16*) alloc((size_t)MROWS * H2 * 2);
  bf16*  oWb    = (bf16*) alloc((size_t)NCLS * H2 * 2);

  // Stage 1: collapse attention to a single per-batch context vector.
  venc_kernel<<<4, 256, 0, stream>>>(a1W, a2W, venc);
  score_kernel<<<BSZ * TENC, 256, 0, stream>>>(encout, venc, scores);
  ctx_kernel<<<BSZ, 256, 0, stream>>>(encout, scores, ctx);
  pre0_kernel<<<512, 256, 0, stream>>>(ctx, Wih0, bih0, bhh0, pre0);
  bias1_kernel<<<16, 256, 0, stream>>>(bih1, bhh1, bias1);

  // Stage 2: bf16 packing of all GEMM operands.
  cvt_kernel<<<1024, 256, 0, stream>>>(Whh0, Whh0b, 2L * G4H * HID);
  build_wcat1_kernel<<<2048, 256, 0, stream>>>(Wih1, Whh1, Wcat1b);
  build_we0_kernel<<<1024, 256, 0, stream>>>(Wih0, We0b);
  gather_emb_kernel<<<1024, 256, 0, stream>>>(input, emb, Xe);
  cvt_kernel<<<4096, 256, 0, stream>>>(outW, oWb, (long)NCLS * H2);
  init_state_kernel<<<(BSZ * H2 + 255) / 256, 256, 0, stream>>>(enc_h0, enc_c0, x1a, h1a, c0, c1);

  // Stage 3: one-shot embedding-gate GEMM for all 64 steps (WMMA).
  xe_gemm_kernel<<<dim3(G4H / 64, MROWS / 128, 2), 256, 0, stream>>>(Xe, We0b, xeg);

  // Stage 4: 64 recurrent steps, each a pair of fused WMMA GEMM + LSTM kernels.
  bf16 *x1c = x1a, *x1n = x1b, *h1c = h1a, *h1n = h1b;
  for (int t = 0; t < TDEC; ++t) {
    step_l0_kernel<<<32, 128, 0, stream>>>(x1c, x1n, c0, Whh0b, xeg, pre0, t);
    step_l1_kernel<<<32, 128, 0, stream>>>(x1n, h1c, h1n, c1, Wcat1b, bias1, decb, t);
    bf16* tmp = x1c; x1c = x1n; x1n = tmp;
    tmp = h1c; h1c = h1n; h1n = tmp;
  }

  // Stage 5: big output projection (134 GFLOP, bf16 WMMA + async-LDS staged weights).
  out_gemm_kernel<<<dim3(NCLS / 64, MROWS / 128), 256, 0, stream>>>(decb, oWb, outb, out);
}
